// SimpleGATv2Layer_48722109006010
// MI455X (gfx1250) — matrix operand order repaired and checked by reading, not compiled, not run on hardware
//
#include <hip/hip_runtime.h>
#include <math.h>

#define N_NODES 20000
#define N_EDGES 320000
#define HIDDEN  64
#define HEADS   4
#define D_HC    256   /* HEADS*HIDDEN == D_INNER */
#define NEG_SLOPE 0.2f
#define BN_EPS  1e-5f

typedef __attribute__((ext_vector_type(2)))  float    v2f;
typedef __attribute__((ext_vector_type(8)))  float    v8f;
typedef __attribute__((ext_vector_type(16))) _Float16 v16h;

#if defined(__HIP_DEVICE_COMPILE__) && __has_builtin(__builtin_amdgcn_wmma_f32_16x16x4_f32)
#define USE_WMMA_F32 1
#else
#define USE_WMMA_F32 0
#endif

// ---- order-preserving float<->int encoding for atomicMax on signed int ----
__device__ __forceinline__ int fenc(float f) {
  int i = __float_as_int(f);
  return i >= 0 ? i : (i ^ 0x7FFFFFFF);
}
__device__ __forceinline__ float fdec(int i) {
  return __int_as_float(i >= 0 ? i : (i ^ 0x7FFFFFFF));
}
#define ENC_NEG_INF ((int)0x807FFFFF)   /* fenc(-inf) */

// ---------------------------------------------------------------- init -----
__global__ void k_init(float* agg, float* denom, int* menc, float* stats,
                       int n_agg, int n_den, int n_stats) {
  int i = blockIdx.x * blockDim.x + threadIdx.x;
  int stride = gridDim.x * blockDim.x;
  for (int j = i; j < n_agg; j += stride) agg[j] = 0.f;
  for (int j = i; j < n_den; j += stride) { denom[j] = 0.f; menc[j] = ENC_NEG_INF; }
  for (int j = i; j < n_stats; j += stride) stats[j] = 0.f;
}

// ---------------------------------------------------- WMMA GEMM kernel -----
// C[M,Nc] = act(A[M,K] @ B[K,Nc] + bias) (+ resid). One wave per 16x16 tile.
// M, K, Nc all multiples of 16 here (20000, 64/256, 256/64).
__global__ void k_gemm(const float* __restrict__ A, const float* __restrict__ B,
                       const float* __restrict__ bias, const float* __restrict__ resid,
                       float* __restrict__ C, int M, int K, int Nc, int do_relu) {
  const int lane = threadIdx.x & 31;
  const int wave = (blockIdx.x * blockDim.x + threadIdx.x) >> 5;
  const int tiles_n = Nc >> 4;
  const int tiles_m = M >> 4;
  if (wave >= tiles_m * tiles_n) return;      // wave-uniform: EXEC stays all-ones
  const int mt   = wave / tiles_n;
  const int nt   = wave % tiles_n;
  const int half = lane >> 4;                 // 0 or 1
  const int l16  = lane & 15;
  const int arow = mt * 16 + l16;
  const int bcol = nt * 16 + l16;

  v8f acc = {};

#if USE_WMMA_F32
  // Full-precision path: V_WMMA_F32_16X16X4_F32, 16 (or 64) MACs along K.
  for (int kk = 0; kk < K; kk += 4) {
    const float* ap = A + (size_t)arow * K + kk + 2 * half;  // A: VGPR j = K=kk+2*half+j
    v2f a; a.x = ap[0]; a.y = ap[1];
    const int bk = kk + 2 * half;                            // B: VGPR j = row kk+2*half+j
    v2f b;
    b.x = B[(size_t)bk * Nc + bcol];
    b.y = B[(size_t)(bk + 1) * Nc + bcol];
    acc = __builtin_amdgcn_wmma_f32_16x16x4_f32(false, a, false, b,
                                                (short)0, acc, false, false);
  }
#else
  // Confirmed-codegen fallback: fp32->fp16 convert, V_WMMA_F32_16X16X32_F16.
  for (int kk = 0; kk < K; kk += 32) {
    v16h a, b;
    const float* ar = A + (size_t)arow * K;
#pragma unroll
    for (int j = 0; j < 4; ++j) {            // VGPRs 0..3: K = kk + 8*half + [0..7]
      int k = kk + 8 * half + 2 * j;
      a[2 * j]     = (_Float16)ar[k];
      a[2 * j + 1] = (_Float16)ar[k + 1];
      int k2 = kk + 16 + 8 * half + 2 * j;   // VGPRs 4..7: K = kk+16+8*half+[0..7]
      a[2 * j + 8] = (_Float16)ar[k2];
      a[2 * j + 9] = (_Float16)ar[k2 + 1];
    }
#pragma unroll
    for (int j = 0; j < 8; ++j) {            // B VGPR j: rows kk+8*half+2j(+8 if j>=4), +1
      int k = kk + 8 * half + 2 * j + (j >= 4 ? 8 : 0);
      b[2 * j]     = (_Float16)B[(size_t)k * Nc + bcol];
      b[2 * j + 1] = (_Float16)B[(size_t)(k + 1) * Nc + bcol];
    }
    acc = __builtin_amdgcn_wmma_f32_16x16x32_f16(false, a, false, b,
                                                 (short)0, acc, false, false);
  }
#endif

  const float bv = bias ? bias[nt * 16 + l16] : 0.f;
#pragma unroll
  for (int i = 0; i < 8; ++i) {              // C/D: VGPR i -> row 8*half + i
    int row = mt * 16 + 8 * half + i;
    size_t off = (size_t)row * Nc + nt * 16 + l16;
    float v = acc[i] + bv;
    if (do_relu) v = v > 0.f ? v : 0.f;
    if (resid) v += resid[off];
    C[off] = v;
  }
}

// ------------------------------------------- edge pass 1: score + max ------
__global__ void k_score(const float* __restrict__ xl, const float* __restrict__ xr,
                        const float* __restrict__ att, const long long* __restrict__ ei,
                        float* __restrict__ score, int* __restrict__ menc) {
  int idx = blockIdx.x * blockDim.x + threadIdx.x;
  if (idx >= N_EDGES * HEADS) return;
  int e = idx >> 2, h = idx & 3;
  int s = (int)ei[e];
  int d = (int)ei[N_EDGES + e];
  const float4* pl = (const float4*)(xl + (size_t)s * D_HC + h * HIDDEN);
  const float4* pr = (const float4*)(xr + (size_t)d * D_HC + h * HIDDEN);
  const float4* pa = (const float4*)(att + h * HIDDEN);
  float acc = 0.f;
#pragma unroll
  for (int q = 0; q < HIDDEN / 4; ++q) {
    float4 a = pl[q], b = pr[q], w = pa[q];
    float v;
    v = a.x + b.x; v = v > 0.f ? v : v * NEG_SLOPE; acc += w.x * v;
    v = a.y + b.y; v = v > 0.f ? v : v * NEG_SLOPE; acc += w.y * v;
    v = a.z + b.z; v = v > 0.f ? v : v * NEG_SLOPE; acc += w.z * v;
    v = a.w + b.w; v = v > 0.f ? v : v * NEG_SLOPE; acc += w.w * v;
  }
  score[idx] = acc;
  atomicMax(menc + d * HEADS + h, fenc(acc));
}

// -------------------------------------- edge pass 2: exp + denom sum -------
__global__ void k_expsum(const long long* __restrict__ ei, const int* __restrict__ menc,
                         float* __restrict__ score, float* __restrict__ denom) {
  int idx = blockIdx.x * blockDim.x + threadIdx.x;
  if (idx >= N_EDGES * HEADS) return;
  int e = idx >> 2, h = idx & 3;
  int d = (int)ei[N_EDGES + e];
  float ex = expf(score[idx] - fdec(menc[d * HEADS + h]));
  score[idx] = ex;                       // reuse buffer: now holds unnormalized alpha
  atomicAdd(denom + d * HEADS + h, ex);
}

// -------------------------------------- edge pass 3: weighted scatter ------
__global__ void k_agg(const long long* __restrict__ ei, const float* __restrict__ score,
                      const float* __restrict__ xl, float* __restrict__ agg) {
  int idx = blockIdx.x * blockDim.x + threadIdx.x;
  if (idx >= N_EDGES * HEADS) return;
  int e = idx >> 2, h = idx & 3;
  int s = (int)ei[e];
  int d = (int)ei[N_EDGES + e];
  float w = score[idx];
  const float* srcp = xl + (size_t)s * D_HC + h * HIDDEN;
  float* dstp = agg + (size_t)d * D_HC + h * HIDDEN;
#pragma unroll 8
  for (int c = 0; c < HIDDEN; ++c) atomicAdd(dstp + c, w * srcp[c]);
}

// --------------------------- node finish: normalize, head-mean, residual ---
__global__ void k_node_finish(const float* __restrict__ x, const float* __restrict__ agg,
                              const float* __restrict__ denom, const float* __restrict__ bias_gat,
                              float* __restrict__ x1) {
  int idx = blockIdx.x * blockDim.x + threadIdx.x;
  if (idx >= N_NODES * HIDDEN) return;
  int n = idx >> 6, c = idx & 63;
  float v = 0.f;
#pragma unroll
  for (int h = 0; h < HEADS; ++h) {
    float dd = fmaxf(denom[n * HEADS + h], 1e-16f);
    v += agg[(size_t)n * D_HC + h * HIDDEN + c] / dd;
  }
  x1[idx] = v * (1.f / HEADS) + bias_gat[c] + x[idx];
}

// ----------------------------------------- BatchNorm stats (sum, sumsq) ----
__global__ void k_bn_stats(const float* __restrict__ src, float* __restrict__ sum,
                           float* __restrict__ sumsq) {
  __shared__ float ss[HIDDEN];
  __shared__ float sq[HIDDEN];
  int c = threadIdx.x & 63, r = threadIdx.x >> 6;  // 4 node-rows per block
  float ps = 0.f, pq = 0.f;
  for (int n = blockIdx.x * 4 + r; n < N_NODES; n += gridDim.x * 4) {
    float v = src[(size_t)n * HIDDEN + c];
    ps += v; pq += v * v;
  }
  if (threadIdx.x < HIDDEN) { ss[c] = 0.f; sq[c] = 0.f; }
  __syncthreads();
  atomicAdd(ss + c, ps);
  atomicAdd(sq + c, pq);
  __syncthreads();
  if (threadIdx.x < HIDDEN) {
    atomicAdd(sum + c, ss[c]);
    atomicAdd(sumsq + c, sq[c]);
  }
}

// ------------------------------------------------------- BatchNorm apply ---
__global__ void k_bn_apply(const float* __restrict__ src, const float* __restrict__ sum,
                           const float* __restrict__ sumsq, const float* __restrict__ gamma,
                           const float* __restrict__ beta, float* __restrict__ dst) {
  int idx = blockIdx.x * blockDim.x + threadIdx.x;
  if (idx >= N_NODES * HIDDEN) return;
  int c = idx & 63;
  float mu  = sum[c] * (1.f / N_NODES);
  float var = sumsq[c] * (1.f / N_NODES) - mu * mu;
  dst[idx] = (src[idx] - mu) * rsqrtf(var + BN_EPS) * gamma[c] + beta[c];
}

// ===========================================================================
extern "C" void kernel_launch(void* const* d_in, const int* in_sizes, int n_in,
                              void* d_out, int out_size, void* d_ws, size_t ws_size,
                              hipStream_t stream) {
  (void)in_sizes; (void)n_in; (void)out_size; (void)ws_size;
  const float*     x        = (const float*)d_in[0];
  const long long* ei       = (const long long*)d_in[1];   // int64 [2, E]
  const float*     W_l      = (const float*)d_in[2];
  const float*     b_l      = (const float*)d_in[3];
  const float*     W_r      = (const float*)d_in[4];
  const float*     b_r      = (const float*)d_in[5];
  const float*     att      = (const float*)d_in[6];
  const float*     bias_gat = (const float*)d_in[7];
  const float*     gamma1   = (const float*)d_in[8];
  const float*     beta1    = (const float*)d_in[9];
  const float*     W1       = (const float*)d_in[10];
  const float*     b1       = (const float*)d_in[11];
  const float*     W2       = (const float*)d_in[12];
  const float*     b2       = (const float*)d_in[13];
  const float*     gamma2   = (const float*)d_in[14];
  const float*     beta2    = (const float*)d_in[15];
  float*           out      = (float*)d_out;

  char* ws = (char*)d_ws;
  size_t off = 0;
  auto alloc = [&](size_t nfloat) { float* p = (float*)(ws + off); off += nfloat * sizeof(float); return p; };
  float* xl    = alloc((size_t)N_NODES * D_HC);   // 20.5 MB
  float* xr    = alloc((size_t)N_NODES * D_HC);   // 20.5 MB
  float* agg   = alloc((size_t)N_NODES * D_HC);   // 20.5 MB
  float* ffh   = alloc((size_t)N_NODES * D_HC);   // 20.5 MB
  float* score = alloc((size_t)N_EDGES * HEADS);  // 5.1 MB
  float* denom = alloc((size_t)N_NODES * HEADS);
  int*   menc  = (int*)alloc((size_t)N_NODES * HEADS);
  float* x1    = alloc((size_t)N_NODES * HIDDEN);
  float* xbn   = alloc((size_t)N_NODES * HIDDEN);
  float* x2    = alloc((size_t)N_NODES * HIDDEN);
  float* stats = alloc(256);  // sum1[64] sumsq1[64] sum2[64] sumsq2[64]

  const int T = 256;  // 8 wave32s per block
  const int g_eh   = (N_EDGES * HEADS + T - 1) / T;     // 5000
  const int g_nc   = (N_NODES * HIDDEN + T - 1) / T;    // 5000
  const int g_g256 = ((N_NODES / 16) * (D_HC / 16) * 32 + T - 1) / T;   // 2500
  const int g_g64  = ((N_NODES / 16) * (HIDDEN / 16) * 32 + T - 1) / T; // 625

  k_init<<<4096, T, 0, stream>>>(agg, denom, menc, stats,
                                 N_NODES * D_HC, N_NODES * HEADS, 256);

  // GATv2 projections
  k_gemm<<<g_g256, T, 0, stream>>>(x, W_l, b_l, nullptr, xl, N_NODES, HIDDEN, D_HC, 0);
  k_gemm<<<g_g256, T, 0, stream>>>(x, W_r, b_r, nullptr, xr, N_NODES, HIDDEN, D_HC, 0);

  // attention: score -> segment max -> exp/sum -> weighted scatter
  k_score <<<g_eh, T, 0, stream>>>(xl, xr, att, ei, score, menc);
  k_expsum<<<g_eh, T, 0, stream>>>(ei, menc, score, denom);
  k_agg   <<<g_eh, T, 0, stream>>>(ei, score, xl, agg);

  // head mean + bias + residual, then BN1
  k_node_finish<<<g_nc, T, 0, stream>>>(x, agg, denom, bias_gat, x1);
  k_bn_stats<<<250, T, 0, stream>>>(x1, stats, stats + 64);
  k_bn_apply<<<g_nc, T, 0, stream>>>(x1, stats, stats + 64, gamma1, beta1, xbn);

  // FFN (relu fused in GEMM1, residual fused in GEMM2), then BN2
  k_gemm<<<g_g256, T, 0, stream>>>(xbn, W1, b1, nullptr, ffh, N_NODES, HIDDEN, D_HC, 1);
  k_gemm<<<g_g64,  T, 0, stream>>>(ffh, W2, b2, xbn, x2, N_NODES, D_HC, HIDDEN, 0);
  k_bn_stats<<<250, T, 0, stream>>>(x2, stats + 128, stats + 192);
  k_bn_apply<<<g_nc, T, 0, stream>>>(x2, stats + 128, stats + 192, gamma2, beta2, out);
}
